// Critic_63848983822714
// MI455X (gfx1250) — matrix operand order repaired and checked by reading, not compiled, not run on hardware
//
#include <hip/hip_runtime.h>

#define NGP_TABLE_SIZE (1u << 18)

typedef _Float16 v16h __attribute__((ext_vector_type(16)));
typedef _Float16 v8h  __attribute__((ext_vector_type(8)));
typedef float    v8f  __attribute__((ext_vector_type(8)));

// Single-instruction ReLU: exactly one v_max_num_f32 (no canonicalize pair).
// WMMA f32 outputs are already canonical; sNaN quieting is wasted work.
static __device__ __forceinline__ float relu1(float v) {
  float r;
  asm("v_max_num_f32_e64 %0, %1, 0" : "=v"(r) : "v"(v));
  return r;
}

// ---------------------------------------------------------------------------
// Fragment buffer layout in d_ws (f16):
//   28 fragments, each 32 lanes x 16 halfs (512 f16 = 1KB), lane-major so a
//   lane's 16 values are 32 contiguous bytes (two global_load_b128).
//   frag 0..3   : w_in  B-fragment, N-tile t (K 0..15 real, 16..31 zero-pad)
//   frag 4..27  : hidden B-fragment, id = 4 + layer*8 + kstep*4 + tile
// B-matrix (32x16 f16, wave32) layout: lane L -> n = L&15, half = L>>4;
//   VGPR j element s  -> idx16 = 2j + s; K = 8*half + idx16 (idx16<8), else 16 + 8*half + idx16-8
// ---------------------------------------------------------------------------
__global__ void ngp_prep_weights(const float* __restrict__ w_in,
                                 const float* __restrict__ w_hidden,
                                 _Float16* __restrict__ frags) {
  int e = blockIdx.x * blockDim.x + threadIdx.x;
  const int TOTAL = 28 * 512;
  if (e >= TOTAL) return;
  int f      = e >> 9;        // fragment id
  int within = e & 511;
  int lane   = within >> 4;   // 0..31
  int idx16  = within & 15;   // element within lane (VGPR j, sub s) = 2j+s
  int half   = lane >> 4;
  int nloc   = lane & 15;
  int k      = (idx16 < 8) ? (8 * half + idx16) : (16 + 8 * half + (idx16 - 8));   // K within this 32-wide fragment
  float v;
  if (f < 4) {                      // w_in [16][64], K padded 16..31 -> 0
    int n = f * 16 + nloc;
    v = (k < 16) ? w_in[k * 64 + n] : 0.0f;
  } else {                          // w_hidden [3][64][64]
    int g     = f - 4;
    int layer = g >> 3;
    int kk    = (g >> 2) & 1;
    int tile  = g & 3;
    int n     = tile * 16 + nloc;
    v = w_hidden[((layer * 64) + (kk * 32 + k)) * 64 + n];
  }
  frags[e] = (_Float16)v;
}

// ---------------------------------------------------------------------------
// Fused hashgrid-encode + 5-layer MLP. One wave32 handles 16 points.
//   lane L: point m = L&15, half h = L>>4 -> computes levels 4h..4h+3
//   (exactly A-fragment K = 8h..8h+7 in VGPR0-3; VGPR4-7 zero-padded).
// ---------------------------------------------------------------------------
__global__ __launch_bounds__(256) void ngp_fused(
    const float* __restrict__ x,
    const float* __restrict__ mesh_min,
    const float* __restrict__ mesh_max,
    const float* __restrict__ tables,
    const _Float16* __restrict__ frags,
    const float* __restrict__ w_out,
    float* __restrict__ out,
    int N) {
  __shared__ __align__(16) _Float16 act[8][16][64];   // per-wave 16x64 f16 tile

  const int lane = threadIdx.x & 31;
  const int wave = threadIdx.x >> 5;
  const int half = lane >> 4;
  const int m    = lane & 15;

  const int pt  = (blockIdx.x * 8 + wave) * 16 + m;
  const int ptc = pt < N ? pt : N - 1;

  // ---- normalize point to [0,1]^3 -------------------------------------
  const float mn0 = mesh_min[0], mn1 = mesh_min[1], mn2 = mesh_min[2];
  const float s0 = 1.0f / (mesh_max[0] - mn0);
  const float s1 = 1.0f / (mesh_max[1] - mn1);
  const float s2 = 1.0f / (mesh_max[2] - mn2);
  const float ux = (x[3 * ptc + 0] - mn0) * s0;
  const float uy = (x[3 * ptc + 1] - mn1) * s1;
  const float uz = (x[3 * ptc + 2] - mn2) * s2;

  // ---- hash-grid encoding: 4 levels per lane --------------------------
  float acc[8];
#pragma unroll
  for (int i = 0; i < 8; ++i) acc[i] = 0.0f;

#pragma unroll
  for (int li = 0; li < 4; ++li) {
    const int   l    = 4 * half + li;     // level 0..7
    const int   resi = 2 << l;            // base_res=2, scale=2
    const float resf = (float)resi;
    const float px = ux * resf, py = uy * resf, pz = uz * resf;
    const float fx = floorf(px), fy = floorf(py), fz = floorf(pz);
    const int   c0x = (int)fx, c0y = (int)fy, c0z = (int)fz;
    const float wx = px - fx, wy = py - fy, wz = pz - fz;
    const int   r1 = resi + 1;
    const bool  dense = (resi <= 32);     // (res+1)^3 <= 2^18 only for levels 0..4
    float a0 = 0.0f, a1 = 0.0f;
#pragma unroll
    for (int c = 0; c < 8; ++c) {
      const int bx = c & 1, by = (c >> 1) & 1, bz = (c >> 2) & 1;
      const int ix = min(max(c0x + bx, 0), resi);
      const int iy = min(max(c0y + by, 0), resi);
      const int iz = min(max(c0z + bz, 0), resi);
      const unsigned h =
          (unsigned)ix ^ ((unsigned)iy * 2654435761u) ^ ((unsigned)iz * 805459861u);
      const int di  = ix + iy * r1 + iz * r1 * r1;
      const int idx = dense ? di : (int)(h & (NGP_TABLE_SIZE - 1u));
      const float2 f2 =
          *(const float2*)(tables + (((size_t)l << 18) + (size_t)idx) * 2);
      const float wt = (bx ? wx : 1.0f - wx) *
                       (by ? wy : 1.0f - wy) *
                       (bz ? wz : 1.0f - wz);
      a0 = fmaf(wt, f2.x, a0);
      a1 = fmaf(wt, f2.y, a1);
    }
    acc[2 * li]     = a0;
    acc[2 * li + 1] = a1;
  }

  // A-fragment for layer 1: elements 0..7 = this lane's 8 features, 8..15 = 0
  v16h a = {};
#pragma unroll
  for (int e2 = 0; e2 < 8; ++e2) a[e2] = (_Float16)acc[e2];

  // ---- layer 1: enc[16x16(pad32)] @ w_in[32x64] -> relu -> act LDS ----
#pragma unroll
  for (int t = 0; t < 4; ++t) {
    const v16h b = *(const v16h*)(frags + (size_t)t * 512 + (size_t)lane * 16);
    v8f c = {};
    c = __builtin_amdgcn_wmma_f32_16x16x32_f16(false, a, false, b,
                                               (short)0, c, false, false);
#pragma unroll
    for (int r = 0; r < 8; ++r)   // C layout: VGPR r -> M = r + 8*half, N = m
      act[wave][r + 8 * half][t * 16 + m] = (_Float16)relu1(c[r]);
  }

  // ---- 3 hidden layers: act[16x64] @ W[64x64] -> relu -----------------
#pragma unroll
  for (int layer = 0; layer < 3; ++layer) {
    asm volatile("s_wait_dscnt 0x0" ::: "memory");
    // A fragments from LDS (A layout: VGPR0-3 <- K 8h..8h+7, VGPR4-7 <- K+16)
    const v8h lo0 = *(const v8h*)&act[wave][m][ 0 + 8 * half];
    const v8h hi0 = *(const v8h*)&act[wave][m][16 + 8 * half];
    const v8h lo1 = *(const v8h*)&act[wave][m][32 + 8 * half];
    const v8h hi1 = *(const v8h*)&act[wave][m][48 + 8 * half];
    const v16h a0 = __builtin_shufflevector(lo0, hi0, 0,1,2,3,4,5,6,7,
                                            8,9,10,11,12,13,14,15);
    const v16h a1 = __builtin_shufflevector(lo1, hi1, 0,1,2,3,4,5,6,7,
                                            8,9,10,11,12,13,14,15);
    v8f d[4];
#pragma unroll
    for (int t = 0; t < 4; ++t) {
      const _Float16* bp =
          frags + (size_t)(4 + layer * 8 + t) * 512 + (size_t)lane * 16;
      const v16h b0 = *(const v16h*)bp;
      const v16h b1 = *(const v16h*)(bp + 4 * 512);   // kstep=1 fragments
      v8f c = {};
      c = __builtin_amdgcn_wmma_f32_16x16x32_f16(false, a0, false, b0,
                                                 (short)0, c, false, false);
      c = __builtin_amdgcn_wmma_f32_16x16x32_f16(false, a1, false, b1,
                                                 (short)0, c, false, false);
      d[t] = c;
    }
#pragma unroll
    for (int t = 0; t < 4; ++t)
#pragma unroll
      for (int r = 0; r < 8; ++r)
        act[wave][r + 8 * half][t * 16 + m] = (_Float16)relu1(d[t][r]);
  }

  // ---- output layer: act[16x64] @ w_out[64] (half-wave split-K) -------
  asm volatile("s_wait_dscnt 0x0" ::: "memory");
  float partial = 0.0f;
  const int kb = half * 32;
#pragma unroll
  for (int q = 0; q < 4; ++q) {     // 8 f16 acts + 8 f32 weights per step
    const v8h    av = *(const v8h*)&act[wave][m][kb + q * 8];
    const float4 wa = *(const float4*)(w_out + kb + q * 8);
    const float4 wb = *(const float4*)(w_out + kb + q * 8 + 4);
    partial = fmaf((float)av[0], wa.x, partial);
    partial = fmaf((float)av[1], wa.y, partial);
    partial = fmaf((float)av[2], wa.z, partial);
    partial = fmaf((float)av[3], wa.w, partial);
    partial = fmaf((float)av[4], wb.x, partial);
    partial = fmaf((float)av[5], wb.y, partial);
    partial = fmaf((float)av[6], wb.z, partial);
    partial = fmaf((float)av[7], wb.w, partial);
  }
  partial += __shfl_xor(partial, 16, 32);
  if (half == 0 && pt < N) out[pt] = partial;
}

// ---------------------------------------------------------------------------
extern "C" void kernel_launch(void* const* d_in, const int* in_sizes, int n_in,
                              void* d_out, int out_size, void* d_ws, size_t ws_size,
                              hipStream_t stream) {
  (void)n_in; (void)out_size; (void)ws_size;
  const float* x        = (const float*)d_in[0];
  const float* mesh_min = (const float*)d_in[1];
  const float* mesh_max = (const float*)d_in[2];
  const float* tables   = (const float*)d_in[3];
  const float* w_in     = (const float*)d_in[4];
  const float* w_hidden = (const float*)d_in[5];
  const float* w_out    = (const float*)d_in[6];
  float* out = (float*)d_out;
  const int N = in_sizes[0] / 3;

  _Float16* frags = (_Float16*)d_ws;   // 28 KB of pre-swizzled f16 B-fragments
  ngp_prep_weights<<<(28 * 512 + 255) / 256, 256, 0, stream>>>(w_in, w_hidden, frags);

  const int blocks = (N + 127) / 128;  // 8 waves/block x 16 points/wave
  ngp_fused<<<blocks, 256, 0, stream>>>(x, mesh_min, mesh_max, tables,
                                        frags, w_out, out, N);
}
